// GateDependentLSTMReLUModelw2Q_37323265803124
// MI455X (gfx1250) — compile-verified, matrix operand order checked by reading, not run on hardware
//
#include <hip/hip_runtime.h>

// Problem constants (match reference)
#define B_  256
#define T_  32
#define Q_  16
#define D_  128
#define H_  128
#define L_  64

typedef __attribute__((ext_vector_type(16))) _Float16 v16h;
typedef __attribute__((ext_vector_type(8)))  _Float16 v8h;
typedef __attribute__((ext_vector_type(8)))  float    v8f;

__device__ __forceinline__ float sigmoidf_(float v) {
  return 1.0f / (1.0f + __expf(-v));
}

// Accumulate one 16x16 output tile per gate-block (i,f,g,o) for column-slice jt.
// sSrc: row-major [16][KK] f16 in LDS (A operand).
// W:    row-major [4*KK][KK] f16 in global (B operand; gates = A @ W^T).
// A fragment layout (ISA 16-bit A 16x32): lanes 0-15 hold K=k0+0..7 / k0+16..23,
// lanes 16-31 hold K=k0+8..15 / k0+24..31, M = lane%16. B mirrored, N = lane%16.
template<int KK>
__device__ __forceinline__ void mma_tiles(const _Float16* __restrict__ sSrc,
                                          const _Float16* __restrict__ W,
                                          int jt, int lane,
                                          v8f& aI, v8f& aF, v8f& aG, v8f& aO)
{
  constexpr int NJT = KK / 16;          // # of 16-col tiles per gate block
  const int m  = lane & 15;
  const int hi = lane >> 4;
  #pragma unroll
  for (int k0 = 0; k0 < KK; k0 += 32) {
    const int base = k0 + hi * 8;
    const _Float16* ap = sSrc + m * KK + base;
    v8h alo = *(const v8h*)(ap);
    v8h ahi = *(const v8h*)(ap + 16);
    v16h a;
    #pragma unroll
    for (int e = 0; e < 8; ++e) { a[e] = alo[e]; a[e + 8] = ahi[e]; }
    #pragma unroll
    for (int gs = 0; gs < 4; ++gs) {
      const int col = (gs * NJT + jt) * 16 + m;     // output column (N) this lane owns
      const _Float16* bp = W + col * KK + base;
      v8h blo = *(const v8h*)(bp);
      v8h bhi = *(const v8h*)(bp + 16);
      v16h bb;
      #pragma unroll
      for (int e = 0; e < 8; ++e) { bb[e] = blo[e]; bb[e + 8] = bhi[e]; }
      v8f& acc = (gs == 0) ? aI : (gs == 1) ? aF : (gs == 2) ? aG : aO;
      acc = __builtin_amdgcn_wmma_f32_16x16x32_f16(
          /*neg_a=*/false, a, /*neg_b=*/false, bb,
          /*c_mod=*/(short)0, acc, /*reuse_a=*/false, /*reuse_b=*/false);
    }
  }
}

// Two-layer LSTM cell for 16 batch rows. TWOQ=0: H=128 (8 active waves);
// TWOQ=1: concatenated 2-qubit state, H=256 (16 active waves).
template<int TWOQ>
__device__ __forceinline__ void gate_impl(
    const float* __restrict__ x,
    const _Float16* __restrict__ Wih0, const _Float16* __restrict__ Whh0,
    const _Float16* __restrict__ Wih1, const _Float16* __restrict__ Whh1,
    const float* __restrict__ bih0, const float* __restrict__ bhh0,
    const float* __restrict__ bih1, const float* __restrict__ bhh1,
    int t, int qa, int qc,
    float* __restrict__ hstate, float* __restrict__ cstate,
    _Float16* sA, _Float16* sH)
{
  constexpr int KK  = TWOQ ? 256 : 128;
  constexpr int NJT = KK / 16;
  const int tid  = threadIdx.x;
  const int lane = tid & 31;
  const int w    = tid >> 5;            // wave id, 0..15 (wave32)
  const int row0 = blockIdx.x * 16;     // this block's batch rows

  // Cooperative load: sA <- x[:, t, q, :], sH <- h_prev layer0 (f32 -> f16)
  for (int idx = tid; idx < 16 * KK; idx += 512) {
    const int rl = idx / KK, d = idx - rl * KK;
    const int b  = row0 + rl;
    const int q  = (TWOQ && d >= H_) ? qc : qa;
    const int dd = (TWOQ && d >= H_) ? d - H_ : d;
    sA[idx] = (_Float16)x[((b * T_ + t) * Q_ + q) * D_ + dd];
    sH[idx] = (_Float16)hstate[((q * 2 + 0) * B_ + b) * H_ + dd];
  }
  __syncthreads();

  #pragma unroll
  for (int layer = 0; layer < 2; ++layer) {
    const _Float16* Wih = layer ? Wih1 : Wih0;
    const _Float16* Whh = layer ? Whh1 : Whh0;
    const float*    bih = layer ? bih1 : bih0;
    const float*    bhh = layer ? bhh1 : bhh0;
    const bool active = (w < NJT);

    v8f aI = {}, aF = {}, aG = {}, aO = {};
    if (active) {
      mma_tiles<KK>(sA, Wih, w, lane, aI, aF, aG, aO);   // input contribution
      mma_tiles<KK>(sH, Whh, w, lane, aI, aF, aG, aO);   // hidden contribution
    }
    __syncthreads();   // all GEMM reads of sA/sH complete

    // Prefetch next layer's h_prev into sH while cells compute
    if (layer == 0) {
      for (int idx = tid; idx < 16 * KK; idx += 512) {
        const int rl = idx / KK, d = idx - rl * KK;
        const int b  = row0 + rl;
        const int q  = (TWOQ && d >= H_) ? qc : qa;
        const int dd = (TWOQ && d >= H_) ? d - H_ : d;
        sH[idx] = (_Float16)hstate[((q * 2 + 1) * B_ + b) * H_ + dd];
      }
    }

    if (active) {
      const int n  = lane & 15;
      const int hi = lane >> 4;
      const int j  = w * 16 + n;                         // hidden index 0..KK-1
      const int q  = (TWOQ && j >= H_) ? qc : qa;
      const int hh = (TWOQ && j >= H_) ? j - H_ : j;
      const float bI = bih[0 * KK + j] + bhh[0 * KK + j];
      const float bF = bih[1 * KK + j] + bhh[1 * KK + j];
      const float bG = bih[2 * KK + j] + bhh[2 * KK + j];
      const float bO = bih[3 * KK + j] + bhh[3 * KK + j];
      #pragma unroll
      for (int r = 0; r < 8; ++r) {
        const int rl = r + hi * 8;                       // local row (C layout: M=r+8*hi)
        const int b  = row0 + rl;
        const int idx = ((q * 2 + layer) * B_ + b) * H_ + hh;
        const float iv = sigmoidf_(aI[r] + bI);
        const float fv = sigmoidf_(aF[r] + bF);
        const float gv = tanhf(aG[r] + bG);
        const float ov = sigmoidf_(aO[r] + bO);
        const float cold = cstate[idx];
        const float cnew = fv * cold + iv * gv;
        const float hnew = ov * tanhf(cnew);
        cstate[idx] = cnew;
        hstate[idx] = hnew;
        if (layer == 0) sA[rl * KK + j] = (_Float16)hnew; // layer1 input
      }
    }
    __syncthreads();   // sA (h0_new) + sH (layer1 h_prev) ready
  }
}

// F16 weight workspace element offsets
#define SQ_W   (5 * 4 * H_ * D_)        /* 327680 per sq weight tensor */
#define TQ_W   (8 * H_ * 2 * H_)        /* 262144 per tq weight tensor */
#define O_SQ_WIH0 0
#define O_SQ_WHH0 (SQ_W)
#define O_SQ_WIH1 (2 * SQ_W)
#define O_SQ_WHH1 (3 * SQ_W)
#define O_TQ_WIH0 (4 * SQ_W)
#define O_TQ_WHH0 (4 * SQ_W + TQ_W)
#define O_TQ_WIH1 (4 * SQ_W + 2 * TQ_W)
#define O_TQ_WHH1 (4 * SQ_W + 3 * TQ_W)

__global__ __launch_bounds__(512)
void gate_step_kernel(const float* __restrict__ x,
                      const _Float16* __restrict__ w16,
                      const float* __restrict__ sq_bih0, const float* __restrict__ sq_bhh0,
                      const float* __restrict__ sq_bih1, const float* __restrict__ sq_bhh1,
                      const float* __restrict__ tq_bih0, const float* __restrict__ tq_bhh0,
                      const float* __restrict__ tq_bih1, const float* __restrict__ tq_bhh1,
                      const int* __restrict__ gate_ids, const int* __restrict__ t_idx,
                      const int* __restrict__ q1_idx, const int* __restrict__ q2_idx,
                      float* __restrict__ hstate, float* __restrict__ cstate, int s)
{
  __shared__ __align__(16) _Float16 sA[16 * 256];
  __shared__ __align__(16) _Float16 sH[16 * 256];
  const int g  = gate_ids[s];
  const int t  = t_idx[s];
  const int qa = q1_idx[s];
  const int qc = q2_idx[s];

  if (g == 5) {
    gate_impl<1>(x,
                 w16 + O_TQ_WIH0, w16 + O_TQ_WHH0, w16 + O_TQ_WIH1, w16 + O_TQ_WHH1,
                 tq_bih0, tq_bhh0, tq_bih1, tq_bhh1,
                 t, qa, qc, hstate, cstate, sA, sH);
  } else {
    const int wo = g * (4 * H_ * D_);   // 65536 per gate type
    const int bo = g * (4 * H_);
    gate_impl<0>(x,
                 w16 + O_SQ_WIH0 + wo, w16 + O_SQ_WHH0 + wo,
                 w16 + O_SQ_WIH1 + wo, w16 + O_SQ_WHH1 + wo,
                 sq_bih0 + bo, sq_bhh0 + bo, sq_bih1 + bo, sq_bhh1 + bo,
                 t, qa, qa, hstate, cstate, sA, sH);
  }
}

__global__ void cvt_f32_to_f16_kernel(const float* __restrict__ src,
                                      _Float16* __restrict__ dst, int n)
{
  const int i = blockIdx.x * blockDim.x + threadIdx.x;
  if (i < n) dst[i] = (_Float16)src[i];
}

__global__ void relu_out_kernel(const float* __restrict__ hstate,
                                float* __restrict__ out)
{
  const int i = blockIdx.x * blockDim.x + threadIdx.x;  // B*Q*H = 524288
  if (i < B_ * Q_ * H_) {
    const int hh = i & (H_ - 1);
    const int q  = (i >> 7) & (Q_ - 1);
    const int b  = i >> 11;
    const float v = hstate[((q * 2 + 1) * B_ + b) * H_ + hh];
    out[i] = v > 0.0f ? v : 0.0f;
  }
}

extern "C" void kernel_launch(void* const* d_in, const int* in_sizes, int n_in,
                              void* d_out, int out_size, void* d_ws, size_t ws_size,
                              hipStream_t stream) {
  const float* x        = (const float*)d_in[0];
  const float* sq_Wih0  = (const float*)d_in[1];
  const float* sq_Whh0  = (const float*)d_in[2];
  const float* sq_bih0  = (const float*)d_in[3];
  const float* sq_bhh0  = (const float*)d_in[4];
  const float* sq_Wih1  = (const float*)d_in[5];
  const float* sq_Whh1  = (const float*)d_in[6];
  const float* sq_bih1  = (const float*)d_in[7];
  const float* sq_bhh1  = (const float*)d_in[8];
  const float* tq_Wih0  = (const float*)d_in[9];
  const float* tq_Whh0  = (const float*)d_in[10];
  const float* tq_bih0  = (const float*)d_in[11];
  const float* tq_bhh0  = (const float*)d_in[12];
  const float* tq_Wih1  = (const float*)d_in[13];
  const float* tq_Whh1  = (const float*)d_in[14];
  const float* tq_bih1  = (const float*)d_in[15];
  const float* tq_bhh1  = (const float*)d_in[16];
  const int*   gate_ids = (const int*)d_in[17];
  const int*   t_idx    = (const int*)d_in[18];
  const int*   q1_idx   = (const int*)d_in[19];
  const int*   q2_idx   = (const int*)d_in[20];

  // Workspace: [hstate f32 Q*2*B*H][cstate f32 Q*2*B*H][f16 weights]
  const size_t STATE_ELEMS = (size_t)Q_ * 2 * B_ * H_;  // 1,048,576
  float*     hstate = (float*)d_ws;
  float*     cstate = hstate + STATE_ELEMS;
  _Float16*  w16    = (_Float16*)(cstate + STATE_ELEMS);

  // Zero initial hidden/cell state (reference: hidden=None -> zeros)
  hipMemsetAsync(d_ws, 0, 2 * STATE_ELEMS * sizeof(float), stream);

  // One-time (per call) weight conversion f32 -> f16
  {
    const float*    srcs[8] = { sq_Wih0, sq_Whh0, sq_Wih1, sq_Whh1,
                                tq_Wih0, tq_Whh0, tq_Wih1, tq_Whh1 };
    const int       offs[8] = { O_SQ_WIH0, O_SQ_WHH0, O_SQ_WIH1, O_SQ_WHH1,
                                O_TQ_WIH0, O_TQ_WHH0, O_TQ_WIH1, O_TQ_WHH1 };
    const int       lens[8] = { SQ_W, SQ_W, SQ_W, SQ_W, TQ_W, TQ_W, TQ_W, TQ_W };
    for (int i = 0; i < 8; ++i) {
      cvt_f32_to_f16_kernel<<<(lens[i] + 255) / 256, 256, 0, stream>>>(
          srcs[i], w16 + offs[i], lens[i]);
    }
  }

  // Sequential circuit: one fused 2-layer LSTM-cell kernel per gate
  for (int s = 0; s < L_; ++s) {
    gate_step_kernel<<<B_ / 16, 512, 0, stream>>>(
        x, w16,
        sq_bih0, sq_bhh0, sq_bih1, sq_bhh1,
        tq_bih0, tq_bhh0, tq_bih1, tq_bhh1,
        gate_ids, t_idx, q1_idx, q2_idx,
        hstate, cstate, s);
  }

  // out[b][q][h] = relu(h_layer1[q][b][h])
  relu_out_kernel<<<(B_ * Q_ * H_ + 255) / 256, 256, 0, stream>>>(hstate, (float*)d_out);
}
